// InstrumentQuantizer_68616397520939
// MI455X (gfx1250) — matrix operand order repaired and checked by reading, not compiled, not run on hardware
//
#include <hip/hip_runtime.h>

// ---------------- constants (match reference) ----------------
constexpr int kNT   = 32768;          // 16 * 2048 tokens
constexpr int kDIM  = 512;
constexpr int kOPD  = 128;
constexpr int kCLS  = 512;
constexpr int kSEQ  = 5;
constexpr int kMMAX = kNT * kSEQ;     // max decoder rows
constexpr float kTAUinv = 0.1f;       // 1/TAU

// ---------------- WMMA types ----------------
typedef __attribute__((ext_vector_type(16))) __bf16 v16bf;
typedef __attribute__((ext_vector_type(8)))  float  v8f;
typedef __attribute__((ext_vector_type(4)))  unsigned int v4u;

union Frag { v4u q[2]; v16bf v; };

__device__ __forceinline__ unsigned short f2bf(float f) {
  unsigned u = __builtin_bit_cast(unsigned, f);
  u = (u + 0x7FFFu + ((u >> 16) & 1u)) >> 16;
  return (unsigned short)u;
}
__device__ __forceinline__ float bf2f(unsigned short s) {
  unsigned u = ((unsigned)s) << 16;
  return __builtin_bit_cast(float, u);
}
__device__ __forceinline__ float geluf(float x) {
  return 0.5f * x * (1.f + tanhf(0.7978845608f * (x + 0.044715f * x * x * x)));
}
__device__ __forceinline__ float gumbelf(int n, int c, int t) {
  unsigned a = (unsigned)(n * 512 + c) * 2654435761u + (unsigned)t * 0x9E3779B9u + 0x85ebca6bu;
  a ^= a >> 16; a *= 0x7feb352du; a ^= a >> 15; a *= 0x846ca68bu; a ^= a >> 16;
  float u = (a >> 8) * (1.f / 16777216.f);
  u = fmaxf(u, 1e-9f);
  return -__logf(-__logf(u));
}

// issue one 16-byte async global->LDS copy (ASYNCcnt-tracked)
__device__ __forceinline__ void async_copy16(const unsigned short* g, const unsigned short* l) {
  unsigned lds = (unsigned)(uintptr_t)l;   // low 32 bits of generic LDS ptr = LDS byte addr
  asm volatile("global_load_async_to_lds_b128 %0, %1, off" :: "v"(lds), "v"(g) : "memory");
}
// stage an NROWSx32 bf16 tile (rows row0.., cols k0..k0+31) into LDS,
// LDS row stride = 40 halfwords (80B, bank-conflict-free for b128 frag reads)
template <int NROWS>
__device__ __forceinline__ void stage_tile(const unsigned short* __restrict__ G, int ldg,
                                           unsigned short* lbase, int row0, int k0, int tid) {
#pragma unroll
  for (int it = 0; it < NROWS / 64; ++it) {
    int ch  = tid + it * 256;             // NROWS*4 chunks of 16B
    int row = ch >> 2, cc = ch & 3;
    async_copy16(G + (size_t)(row0 + row) * ldg + k0 + cc * 8,
                 lbase + row * 40 + cc * 8);
  }
}

// =====================================================================
// bf16 WMMA GEMM with async-LDS double buffering:
//   D[M,N] = A[M,K] @ BT[N,K]^T (+bias)(+residual)
// block = 128x128, 8 waves, wave tile 64x32 (8 x v_wmma_f32_16x16x32_bf16)
// =====================================================================
__global__ __launch_bounds__(256)
void gemm_bf16_async_kernel(const unsigned short* __restrict__ A,
                            const unsigned short* __restrict__ BT,
                            const float* __restrict__ bias,
                            const float* __restrict__ residual,
                            float* __restrict__ Df,
                            unsigned short* __restrict__ Dbf,
                            int M, int N, int K) {
  (void)M;
  __shared__ __align__(16) unsigned short sA[2][128 * 40];
  __shared__ __align__(16) unsigned short sB[2][128 * 40];
  const int tid  = threadIdx.x;
  const int lane = tid & 31;
  const int wave = tid >> 5;
  const int wr = wave >> 2;             // 0..1 -> 64-row group
  const int wc = wave & 3;              // 0..3 -> 32-col group
  const int rowB = blockIdx.y * 128;
  const int colB = blockIdx.x * 128;
  const int half = lane >> 4;
  const int l16  = lane & 15;

  v8f acc[8];
#pragma unroll
  for (int i = 0; i < 8; ++i)
#pragma unroll
    for (int j = 0; j < 8; ++j) acc[i][j] = 0.f;

  // prologue: stage first K-slab into buffer 0 (4 async issues per thread)
  stage_tile<128>(A,  K, sA[0], rowB, 0, tid);
  stage_tile<128>(BT, K, sB[0], colB, 0, tid);

  for (int k0 = 0; k0 < K; k0 += 32) {
    const int cur = (k0 >> 5) & 1, nxt = cur ^ 1;
    if (k0 + 32 < K) {
      // stage next slab into the other buffer (consumed+barriered last iter)
      stage_tile<128>(A,  K, sA[nxt], rowB, k0 + 32, tid);
      stage_tile<128>(BT, K, sB[nxt], colB, k0 + 32, tid);
      // 8 outstanding, in-order: <=4 means current slab has landed
      asm volatile("s_wait_asynccnt 0x4" ::: "memory");
    } else {
      asm volatile("s_wait_asynccnt 0x0" ::: "memory");
    }
    __syncthreads();

    Frag a[4], b[2];
#pragma unroll
    for (int r = 0; r < 4; ++r) {
      const v4u* p = (const v4u*)&sA[cur][(size_t)(wr * 64 + r * 16 + l16) * 40 + half * 16];
      a[r].q[0] = p[0];
      a[r].q[1] = p[1];                 // +8 halfwords
    }
#pragma unroll
    for (int c = 0; c < 2; ++c) {
      const v4u* p = (const v4u*)&sB[cur][(size_t)(wc * 32 + c * 16 + l16) * 40 + half * 16];
      b[c].q[0] = p[0];
      b[c].q[1] = p[1];
    }
#pragma unroll
    for (int r = 0; r < 4; ++r)
#pragma unroll
      for (int c = 0; c < 2; ++c)
        acc[r * 2 + c] = __builtin_amdgcn_wmma_f32_16x16x32_bf16(
            false, a[r].v, false, b[c].v, (short)0, acc[r * 2 + c], false, false);

    __syncthreads();                    // all reads done before buffer reuse
  }

#pragma unroll
  for (int r = 0; r < 4; ++r)
#pragma unroll
    for (int c = 0; c < 2; ++c) {
      const int col = colB + wc * 32 + c * 16 + l16;
      const float bv = bias ? bias[col] : 0.f;
#pragma unroll
      for (int i = 0; i < 8; ++i) {
        const int m = rowB + wr * 64 + r * 16 + half * 8 + i;
        float v = acc[r * 2 + c][i] + bv;
        if (residual) v += residual[(size_t)m * N + col];
        if (Df)  Df[(size_t)m * N + col] = v;
        if (Dbf) Dbf[(size_t)m * N + col] = f2bf(v);
      }
    }
}

// =====================================================================
// Fused GEGLU GEMM (async-LDS): hid = X[M,K] @ W1T[2048,K]^T + b1,
// out = a * gelu(g), out[M,1024] bf16.
// block = 128 rows x 64 'a'-cols, 8 waves, wave tile 32x32 dual (a+gate).
// Stages A (128x32) and both B column tiles (64x32 each) per K-slab.
// =====================================================================
__global__ __launch_bounds__(256)
void gemm_bf16_geglu_kernel(const unsigned short* __restrict__ A,
                            const unsigned short* __restrict__ BT,   // [2048,K]
                            const float* __restrict__ bias,          // [2048]
                            unsigned short* __restrict__ O,          // [M,1024]
                            int M, int K) {
  (void)M;
  __shared__ __align__(16) unsigned short sA[2][128 * 40];
  __shared__ __align__(16) unsigned short sBa[2][64 * 40];
  __shared__ __align__(16) unsigned short sBg[2][64 * 40];
  const int tid  = threadIdx.x;
  const int lane = tid & 31;
  const int wave = tid >> 5;
  const int wr = wave >> 1;             // 0..3 -> 32-row group
  const int wc = wave & 1;              // 0..1 -> 32-col group
  const int rowB = blockIdx.y * 128;
  const int colB = blockIdx.x * 64;     // within [0,1024)
  const int half = lane >> 4;
  const int l16  = lane & 15;

  v8f accA[4], accG[4];
#pragma unroll
  for (int i = 0; i < 4; ++i)
#pragma unroll
    for (int j = 0; j < 8; ++j) { accA[i][j] = 0.f; accG[i][j] = 0.f; }

  // prologue: 2 (A) + 1 (Ba) + 1 (Bg) = 4 async issues per thread
  stage_tile<128>(A,  K, sA[0],  rowB,        0, tid);
  stage_tile<64>(BT,  K, sBa[0], colB,        0, tid);
  stage_tile<64>(BT,  K, sBg[0], colB + 1024, 0, tid);

  for (int k0 = 0; k0 < K; k0 += 32) {
    const int cur = (k0 >> 5) & 1, nxt = cur ^ 1;
    if (k0 + 32 < K) {
      stage_tile<128>(A,  K, sA[nxt],  rowB,        k0 + 32, tid);
      stage_tile<64>(BT,  K, sBa[nxt], colB,        k0 + 32, tid);
      stage_tile<64>(BT,  K, sBg[nxt], colB + 1024, k0 + 32, tid);
      asm volatile("s_wait_asynccnt 0x4" ::: "memory");
    } else {
      asm volatile("s_wait_asynccnt 0x0" ::: "memory");
    }
    __syncthreads();

    Frag a[2], bA[2], bG[2];
#pragma unroll
    for (int r = 0; r < 2; ++r) {
      const v4u* p = (const v4u*)&sA[cur][(size_t)(wr * 32 + r * 16 + l16) * 40 + half * 16];
      a[r].q[0] = p[0]; a[r].q[1] = p[1];
    }
#pragma unroll
    for (int c = 0; c < 2; ++c) {
      const v4u* pa = (const v4u*)&sBa[cur][(size_t)(wc * 32 + c * 16 + l16) * 40 + half * 16];
      bA[c].q[0] = pa[0]; bA[c].q[1] = pa[1];
      const v4u* pg = (const v4u*)&sBg[cur][(size_t)(wc * 32 + c * 16 + l16) * 40 + half * 16];
      bG[c].q[0] = pg[0]; bG[c].q[1] = pg[1];
    }
#pragma unroll
    for (int r = 0; r < 2; ++r)
#pragma unroll
      for (int c = 0; c < 2; ++c) {
        accA[r * 2 + c] = __builtin_amdgcn_wmma_f32_16x16x32_bf16(
            false, a[r].v, false, bA[c].v, (short)0, accA[r * 2 + c], false, false);
        accG[r * 2 + c] = __builtin_amdgcn_wmma_f32_16x16x32_bf16(
            false, a[r].v, false, bG[c].v, (short)0, accG[r * 2 + c], false, false);
      }

    __syncthreads();
  }

#pragma unroll
  for (int r = 0; r < 2; ++r)
#pragma unroll
    for (int c = 0; c < 2; ++c) {
      const int col = colB + wc * 32 + c * 16 + l16;
      const float ba = bias[col], bg = bias[col + 1024];
#pragma unroll
      for (int i = 0; i < 8; ++i) {
        const int m = rowB + wr * 32 + r * 16 + half * 8 + i;
        const float va = accA[r * 2 + c][i] + ba;
        const float vg = accG[r * 2 + c][i] + bg;
        O[(size_t)m * 1024 + col] = f2bf(va * geluf(vg));
      }
    }
}

// ---------------- elementwise / reduction kernels ----------------
__global__ void zero_f32_kernel(float* p, long long n) {
  long long i = (long long)blockIdx.x * 256 + threadIdx.x;
  if (i < n) p[i] = 0.f;
}
__global__ void copy_f32_kernel(const float* s, float* d, long long n) {
  long long i = (long long)blockIdx.x * 256 + threadIdx.x;
  if (i < n) d[i] = s[i];
}
__global__ void permute_kernel(const float* __restrict__ x, float* __restrict__ px,
                               unsigned short* __restrict__ pxb) {
  int i = blockIdx.x * 256 + threadIdx.x;
  if (i >= kNT * kOPD) return;
  int c = i & 127, n = i >> 7, b = n >> 11, s = n & 2047;
  float v = x[((size_t)b * 128 + c) * 2048 + s];
  px[i] = v; pxb[i] = f2bf(v);
}
__global__ void transpose_cast_kernel(const float* __restrict__ w, unsigned short* __restrict__ wT,
                                      int K, int N, int zero_first_row) {
  long long i = (long long)blockIdx.x * 256 + threadIdx.x;
  if (i >= (long long)K * N) return;
  int n = (int)(i / K), k = (int)(i % K);
  float v = (zero_first_row && k == 0) ? 0.f : w[(size_t)k * N + n];
  wT[i] = f2bf(v);
}
__global__ void bn_stats_kernel(const float* __restrict__ h, float* mean, float* rstd, int M) {
  int c = blockIdx.x, tid = threadIdx.x;
  float s = 0.f, s2 = 0.f;
  for (int r = tid; r < M; r += 256) {
    float v = h[(size_t)r * kDIM + c];
    s += v; s2 += v * v;
  }
  __shared__ float rs[256], rq[256];
  rs[tid] = s; rq[tid] = s2; __syncthreads();
  for (int o = 128; o > 0; o >>= 1) {
    if (tid < o) { rs[tid] += rs[tid + o]; rq[tid] += rq[tid + o]; }
    __syncthreads();
  }
  if (tid == 0) {
    float m = rs[0] / M, var = rq[0] / M - m * m;
    mean[c] = m; rstd[c] = rsqrtf(var + 1e-5f);
  }
}
__global__ void bn_apply_kernel(const float* __restrict__ h, const float* mean, const float* rstd,
                                const float* g, const float* b, unsigned short* y, int total) {
  int i = blockIdx.x * 256 + threadIdx.x;
  if (i >= total) return;
  int c = i & (kDIM - 1);
  y[i] = f2bf((h[i] - mean[c]) * rstd[c] * g[c] + b[c]);
}
__global__ void rms_kernel(const float* __restrict__ x, const float* __restrict__ g,
                           unsigned short* __restrict__ y) {
  int row = blockIdx.x, tid = threadIdx.x;
  const float* xr = x + (size_t)row * kDIM;
  float a = xr[tid], b = xr[tid + 256];
  __shared__ float red[256];
  red[tid] = a * a + b * b; __syncthreads();
  for (int o = 128; o > 0; o >>= 1) { if (tid < o) red[tid] += red[tid + o]; __syncthreads(); }
  float sc = rsqrtf(red[0] * (1.f / kDIM) + 1e-8f);
  y[(size_t)row * kDIM + tid]       = f2bf(a * g[tid] * sc);
  y[(size_t)row * kDIM + tid + 256] = f2bf(b * g[tid + 256] * sc);
}
__global__ void stack_to_xcur_kernel(const float* __restrict__ S, float* __restrict__ xc,
                                     int tlen) {
  long long i = (long long)blockIdx.x * 256 + threadIdx.x;
  long long total = (long long)kNT * tlen * kDIM;
  if (i >= total) return;
  int d = (int)(i & (kDIM - 1));
  long long row = i >> 9;
  int j = (int)(row % tlen);
  long long n = row / tlen;
  xc[i] = S[((long long)j * kNT + n) * kDIM + d];
}
// rotary + causal attention: one wave per (token n, position j, head h)
__global__ void attention_kernel(const unsigned short* __restrict__ qb,
                                 const unsigned short* __restrict__ kb,
                                 const unsigned short* __restrict__ vb,
                                 unsigned short* __restrict__ ob, int tlen) {
  int gid = blockIdx.x * blockDim.x + threadIdx.x;
  int w = gid >> 5, lane = gid & 31;
  int total = kNT * tlen * 2;
  if (w >= total) return;
  int hh = w & 1;
  int j = (w >> 1) % tlen;
  long long n = (w >> 1) / tlen;
  size_t rowQ = ((size_t)n * tlen + j) * kDIM + hh * 256;
  __builtin_prefetch(vb + rowQ, 0, 3);            // V rows reused across waves

  float inv[4], cq[4], sq[4], q1[4], q2[4], qr1[4], qr2[4];
#pragma unroll
  for (int u = 0; u < 4; ++u) {
    int p = lane * 4 + u;                         // pair index 0..127
    inv[u] = __expf(-(float)p * 0.0719558f);      // log(1e4)/128
    float ang = (float)j * inv[u];
    cq[u] = __cosf(ang); sq[u] = __sinf(ang);
    q1[u] = bf2f(qb[rowQ + p]);
    q2[u] = bf2f(qb[rowQ + 128 + p]);
    qr1[u] = q1[u] * cq[u] - q2[u] * sq[u];
    qr2[u] = q2[u] * cq[u] + q1[u] * sq[u];
  }
  float sc[8];
  for (int jk = 0; jk <= j; ++jk) {
    size_t rowK = ((size_t)n * tlen + jk) * kDIM + hh * 256;
    float part = 0.f;
#pragma unroll
    for (int u = 0; u < 4; ++u) {
      int p = lane * 4 + u;
      float ang = (float)jk * inv[u];
      float ck = __cosf(ang), sk = __sinf(ang);
      float k1 = bf2f(kb[rowK + p]), k2 = bf2f(kb[rowK + 128 + p]);
      float kr1 = k1 * ck - k2 * sk;
      float kr2 = k2 * ck + k1 * sk;
      part += qr1[u] * kr1 + qr2[u] * kr2;
    }
    for (int o = 16; o > 0; o >>= 1) part += __shfl_xor(part, o, 32);
    sc[jk] = part * 0.0625f;                      // 1/sqrt(256)
  }
  float mx = -1e30f;
  for (int jk = 0; jk <= j; ++jk) mx = fmaxf(mx, sc[jk]);
  float ssum = 0.f;
  for (int jk = 0; jk <= j; ++jk) { sc[jk] = __expf(sc[jk] - mx); ssum += sc[jk]; }
  float isum = 1.f / ssum;
  float o1[4] = {0, 0, 0, 0}, o2[4] = {0, 0, 0, 0};
  for (int jk = 0; jk <= j; ++jk) {
    size_t rowV = ((size_t)n * tlen + jk) * kDIM + hh * 256;
    float wgt = sc[jk] * isum;
#pragma unroll
    for (int u = 0; u < 4; ++u) {
      int p = lane * 4 + u;
      o1[u] += wgt * bf2f(vb[rowV + p]);
      o2[u] += wgt * bf2f(vb[rowV + 128 + p]);
    }
  }
#pragma unroll
  for (int u = 0; u < 4; ++u) {
    int p = lane * 4 + u;
    ob[rowQ + p]       = f2bf(o1[u]);
    ob[rowQ + 128 + p] = f2bf(o2[u]);
  }
}
__global__ void gather_last_kernel(const unsigned short* __restrict__ dh,
                                   unsigned short* __restrict__ last, int tlen) {
  int i = blockIdx.x * 256 + threadIdx.x;
  if (i >= kNT * kDIM) return;
  int d = i & (kDIM - 1), n = i >> 9;
  last[i] = dh[((size_t)n * tlen + tlen - 1) * kDIM + d];
}
// gumbel-softmax sampling, marginal accumulation, argmax codes, c_gs (bf16)
__global__ void sample_kernel(const float* __restrict__ logits, int* __restrict__ codes,
                              int t, float* __restrict__ marginal,
                              unsigned short* __restrict__ cgs, float invN) {
  int n = blockIdx.x, tid = threadIdx.x;
  __shared__ float rf[256];
  __shared__ int   ri[256];
  float l0 = logits[(size_t)n * kCLS + tid];
  float l1 = logits[(size_t)n * kCLS + 256 + tid];
  for (int s = 0; s < t; ++s) {
    int uc = codes[n * kSEQ + s];
    if (uc == tid)       l0 = -1e30f;
    if (uc == tid + 256) l1 = -1e30f;
  }
  // soft = softmax(logits), marginal += soft/N
  rf[tid] = fmaxf(l0, l1); __syncthreads();
  for (int o = 128; o > 0; o >>= 1) { if (tid < o) rf[tid] = fmaxf(rf[tid], rf[tid + o]); __syncthreads(); }
  float mx = rf[0]; __syncthreads();
  float e0 = __expf(l0 - mx), e1 = __expf(l1 - mx);
  rf[tid] = e0 + e1; __syncthreads();
  for (int o = 128; o > 0; o >>= 1) { if (tid < o) rf[tid] += rf[tid + o]; __syncthreads(); }
  float isum = 1.f / rf[0]; __syncthreads();
  atomicAdd(&marginal[tid],       e0 * isum * invN);
  atomicAdd(&marginal[tid + 256], e1 * isum * invN);
  // gumbel-perturbed logits
  float z0 = (l0 <= -1e29f) ? -1e30f : l0 + gumbelf(n, tid, t);
  float z1 = (l1 <= -1e29f) ? -1e30f : l1 + gumbelf(n, tid + 256, t);
  // argmax(z) == argmax(c_gs)
  float zv = (z0 >= z1) ? z0 : z1;
  int   zi = (z0 >= z1) ? tid : tid + 256;
  rf[tid] = zv; ri[tid] = zi; __syncthreads();
  for (int o = 128; o > 0; o >>= 1) {
    if (tid < o && rf[tid + o] > rf[tid]) { rf[tid] = rf[tid + o]; ri[tid] = ri[tid + o]; }
    __syncthreads();
  }
  if (tid == 0) codes[n * kSEQ + t] = ri[0];
  float zmax = rf[0]; __syncthreads();
  float q0 = __expf((z0 - zmax) * kTAUinv);
  float q1 = __expf((z1 - zmax) * kTAUinv);
  rf[tid] = q0 + q1; __syncthreads();
  for (int o = 128; o > 0; o >>= 1) { if (tid < o) rf[tid] += rf[tid + o]; __syncthreads(); }
  float qis = 1.f / rf[0];
  cgs[(size_t)n * kCLS + tid]       = f2bf(q0 * qis);
  cgs[(size_t)n * kCLS + 256 + tid] = f2bf(q1 * qis);
}
__global__ void perp_kernel(const float* __restrict__ marginal, float* __restrict__ out) {
  int tid = threadIdx.x;
  __shared__ float red[256];
  float m0 = marginal[tid], m1 = marginal[tid + 256];
  red[tid] = m0 * __logf(m0 + 1e-7f) + m1 * __logf(m1 + 1e-7f);
  __syncthreads();
  for (int o = 128; o > 0; o >>= 1) { if (tid < o) red[tid] += red[tid + o]; __syncthreads(); }
  if (tid == 0) atomicAdd(&out[5], __expf(-red[0]));
}
__global__ void loss_kernel(const float* __restrict__ recon, const float* __restrict__ px,
                            float* __restrict__ out, int total) {
  __shared__ float red[256];
  float s = 0.f;
  for (int i = blockIdx.x * 256 + threadIdx.x; i < total; i += gridDim.x * 256) {
    float d = recon[i] - px[i];
    s += d * d;
  }
  red[threadIdx.x] = s; __syncthreads();
  for (int o = 128; o > 0; o >>= 1) { if (threadIdx.x < o) red[threadIdx.x] += red[threadIdx.x + o]; __syncthreads(); }
  if (threadIdx.x == 0) atomicAdd(out, red[0]);
}
__global__ void finalize_kernel(float* out, float invElems) {
  int t = threadIdx.x;
  if (t < 5)      out[t] *= ((float)(t + 1) / 5.f) * invElems;
  else if (t == 5) out[5] *= 0.2f;
}

// =====================================================================
extern "C" void kernel_launch(void* const* d_in, const int* in_sizes, int n_in,
                              void* d_out, int out_size, void* d_ws, size_t ws_size,
                              hipStream_t stream) {
  (void)in_sizes; (void)n_in; (void)out_size; (void)ws_size;

  // ---- input pointers (setup_inputs dict order, depth-first) ----
  const float* X      = (const float*)d_in[0];
  const float* proj_w = (const float*)d_in[1];
  const float* proj_b = (const float*)d_in[2];
  const float *enc_w1[4], *enc_b1[4], *enc_w2[4], *enc_b2[4], *enc_g[4], *enc_bb[4];
  for (int i = 0; i < 4; ++i) {
    int b = 3 + 6 * i;
    enc_w1[i] = (const float*)d_in[b + 0]; enc_b1[i] = (const float*)d_in[b + 1];
    enc_w2[i] = (const float*)d_in[b + 2]; enc_b2[i] = (const float*)d_in[b + 3];
    enc_g[i]  = (const float*)d_in[b + 4]; enc_bb[i] = (const float*)d_in[b + 5];
  }
  const float *dw1[2], *db1[2], *dw2[2], *db2[2], *dag[2], *dfg[2], *dwq[2], *dwk[2], *dwv[2], *dwo[2];
  for (int l = 0; l < 2; ++l) {
    int b = 27 + 10 * l;
    dw1[l] = (const float*)d_in[b + 0]; db1[l] = (const float*)d_in[b + 1];
    dw2[l] = (const float*)d_in[b + 2]; db2[l] = (const float*)d_in[b + 3];
    dag[l] = (const float*)d_in[b + 4]; dfg[l] = (const float*)d_in[b + 5];
    dwq[l] = (const float*)d_in[b + 6]; dwk[l] = (const float*)d_in[b + 7];
    dwv[l] = (const float*)d_in[b + 8]; dwo[l] = (const float*)d_in[b + 9];
  }
  const float* final_g = (const float*)d_in[47];
  const float* cls_w   = (const float*)d_in[48];
  const float* cls_b   = (const float*)d_in[49];
  const float* fb_w    = (const float*)d_in[50];
  const float* cb_w    = (const float*)d_in[51];
  float* out = (float*)d_out;

  // ---- workspace allocator ----
  size_t off = 0;
  auto alloc = [&](size_t bytes) -> void* {
    void* p = (char*)d_ws + off;
    off += (bytes + 255) & ~(size_t)255;
    return p;
  };
  float*          px    = (float*)alloc((size_t)kNT * kOPD * 4);
  unsigned short* pxb   = (unsigned short*)alloc((size_t)kNT * kOPD * 2);
  unsigned short* projT = (unsigned short*)alloc((size_t)kDIM * kOPD * 2);
  unsigned short *encW1T[4], *encW2T[4];
  for (int i = 0; i < 4; ++i) {
    encW1T[i] = (unsigned short*)alloc((size_t)2048 * 512 * 2);
    encW2T[i] = (unsigned short*)alloc((size_t)512 * 1024 * 2);
  }
  unsigned short *dQT[2], *dKT[2], *dVT[2], *dOT[2], *dW1T[2], *dW2T[2];
  for (int l = 0; l < 2; ++l) {
    dQT[l]  = (unsigned short*)alloc((size_t)512 * 512 * 2);
    dKT[l]  = (unsigned short*)alloc((size_t)512 * 512 * 2);
    dVT[l]  = (unsigned short*)alloc((size_t)512 * 512 * 2);
    dOT[l]  = (unsigned short*)alloc((size_t)512 * 512 * 2);
    dW1T[l] = (unsigned short*)alloc((size_t)2048 * 512 * 2);
    dW2T[l] = (unsigned short*)alloc((size_t)512 * 1024 * 2);
  }
  unsigned short* clsT = (unsigned short*)alloc((size_t)512 * 512 * 2);
  unsigned short* fbT  = (unsigned short*)alloc((size_t)512 * 512 * 2);
  unsigned short* cbT  = (unsigned short*)alloc((size_t)128 * 512 * 2);
  float* h    = (float*)alloc((size_t)kNT * kDIM * 4);
  float* bnM  = (float*)alloc(512 * 4);
  float* bnR  = (float*)alloc(512 * 4);
  unsigned short* abuf  = (unsigned short*)alloc((size_t)kMMAX * kDIM * 2);
  unsigned short* geglu = (unsigned short*)alloc((size_t)kMMAX * 1024 * 2);
  float* S    = (float*)alloc((size_t)kSEQ * kNT * kDIM * 4);
  float* xcur = (float*)alloc((size_t)kMMAX * kDIM * 4);
  unsigned short* qb  = (unsigned short*)alloc((size_t)kMMAX * kDIM * 2);
  unsigned short* kb  = (unsigned short*)alloc((size_t)kMMAX * kDIM * 2);
  unsigned short* vb  = (unsigned short*)alloc((size_t)kMMAX * kDIM * 2);
  unsigned short* aob = (unsigned short*)alloc((size_t)kMMAX * kDIM * 2);
  unsigned short* dhb = (unsigned short*)alloc((size_t)kMMAX * kDIM * 2);
  unsigned short* lastb = (unsigned short*)alloc((size_t)kNT * kDIM * 2);
  float* logits = (float*)alloc((size_t)kNT * kCLS * 4);
  unsigned short* cgs = (unsigned short*)alloc((size_t)kNT * kCLS * 2);
  int*   codes  = (int*)alloc((size_t)kNT * kSEQ * 4);
  float* marg   = (float*)alloc(512 * 4);
  float* recon  = (float*)alloc((size_t)kNT * kOPD * 4);

  // ---- launchers ----
  auto gemm = [&](const unsigned short* A, const unsigned short* BT, const float* bias,
                  const float* res, float* Df, unsigned short* Dbf, int M, int N, int K) {
    dim3 g(N / 128, M / 128);
    gemm_bf16_async_kernel<<<g, 256, 0, stream>>>(A, BT, bias, res, Df, Dbf, M, N, K);
  };
  auto gemm_geglu = [&](const unsigned short* A, const unsigned short* BT, const float* bias,
                        unsigned short* O, int M, int K) {
    dim3 g(1024 / 64, M / 128);
    gemm_bf16_geglu_kernel<<<g, 256, 0, stream>>>(A, BT, bias, O, M, K);
  };
  auto tcast = [&](const float* w, unsigned short* wT, int K, int N, int zf) {
    long long tot = (long long)K * N;
    transpose_cast_kernel<<<(int)((tot + 255) / 256), 256, 0, stream>>>(w, wT, K, N, zf);
  };
  auto zero = [&](float* p, long long n) {
    zero_f32_kernel<<<(int)((n + 255) / 256), 256, 0, stream>>>(p, n);
  };

  // ---- init ----
  zero(out, 6);
  zero(recon, (long long)kNT * kOPD);
  permute_kernel<<<(kNT * kOPD + 255) / 256, 256, 0, stream>>>(X, px, pxb);

  tcast(proj_w, projT, kOPD, kDIM, 0);
  for (int i = 0; i < 4; ++i) {
    tcast(enc_w1[i], encW1T[i], 512, 2048, 0);
    tcast(enc_w2[i], encW2T[i], 1024, 512, 0);
  }
  for (int l = 0; l < 2; ++l) {
    tcast(dwq[l], dQT[l], 512, 512, 0);
    tcast(dwk[l], dKT[l], 512, 512, 0);
    tcast(dwv[l], dVT[l], 512, 512, 0);
    tcast(dwo[l], dOT[l], 512, 512, 0);
    tcast(dw1[l], dW1T[l], 512, 2048, 0);
    tcast(dw2[l], dW2T[l], 1024, 512, 0);
  }
  tcast(cls_w, clsT, 512, 512, 0);
  tcast(fb_w, fbT, 512, 512, 0);
  tcast(cb_w, cbT, 512, 128, 1);   // codebook row 0 -> 0

  // ---- projection ----
  gemm(pxb, projT, proj_b, nullptr, h, nullptr, kNT, kDIM, kOPD);

  // ---- encoder: 4 x (BN -> GEGLU FF -> residual) ----
  for (int i = 0; i < 4; ++i) {
    bn_stats_kernel<<<512, 256, 0, stream>>>(h, bnM, bnR, kNT);
    bn_apply_kernel<<<(kNT * kDIM + 255) / 256, 256, 0, stream>>>(h, bnM, bnR, enc_g[i], enc_bb[i], abuf, kNT * kDIM);
    gemm_geglu(abuf, encW1T[i], enc_b1[i], geglu, kNT, kDIM);
    gemm(geglu, encW2T[i], enc_b2[i], h, h, nullptr, kNT, kDIM, 1024);
  }
  copy_f32_kernel<<<(int)(((long long)kNT * kDIM + 255) / 256), 256, 0, stream>>>(h, S, (long long)kNT * kDIM);

  // ---- decoder loop ----
  for (int t = 0; t < kSEQ; ++t) {
    const int tlen = t + 1;
    const int M = kNT * tlen;
    stack_to_xcur_kernel<<<(int)(((long long)M * kDIM + 255) / 256), 256, 0, stream>>>(S, xcur, tlen);

    for (int l = 0; l < 2; ++l) {
      rms_kernel<<<M, 256, 0, stream>>>(xcur, dag[l], abuf);
      gemm(abuf, dQT[l], nullptr, nullptr, nullptr, qb, M, kDIM, kDIM);
      gemm(abuf, dKT[l], nullptr, nullptr, nullptr, kb, M, kDIM, kDIM);
      gemm(abuf, dVT[l], nullptr, nullptr, nullptr, vb, M, kDIM, kDIM);
      {
        long long waves = (long long)kNT * tlen * 2;
        int blocks = (int)((waves * 32 + 255) / 256);
        attention_kernel<<<blocks, 256, 0, stream>>>(qb, kb, vb, aob, tlen);
      }
      gemm(aob, dOT[l], nullptr, xcur, xcur, nullptr, M, kDIM, kDIM);
      rms_kernel<<<M, 256, 0, stream>>>(xcur, dfg[l], abuf);
      gemm_geglu(abuf, dW1T[l], db1[l], geglu, M, kDIM);
      gemm(geglu, dW2T[l], db2[l], xcur, xcur, nullptr, M, kDIM, 1024);
    }
    rms_kernel<<<M, 256, 0, stream>>>(xcur, final_g, dhb);
    gather_last_kernel<<<(kNT * kDIM + 255) / 256, 256, 0, stream>>>(dhb, lastb, tlen);
    gemm(lastb, clsT, cls_b, nullptr, logits, nullptr, kNT, kCLS, kDIM);

    zero(marg, 512);
    sample_kernel<<<kNT, 256, 0, stream>>>(logits, codes, t, marg, cgs, 1.f / (float)kNT);
    perp_kernel<<<1, 256, 0, stream>>>(marg, out);

    if (t + 1 < kSEQ)
      gemm(cgs, fbT, nullptr, nullptr, S + (size_t)(t + 1) * kNT * kDIM, nullptr, kNT, kDIM, kCLS);
    gemm(cgs, cbT, nullptr, recon, recon, nullptr, kNT, kOPD, kCLS);
    loss_kernel<<<512, 256, 0, stream>>>(recon, px, out + t, kNT * kOPD);
  }
  finalize_kernel<<<1, 6, 0, stream>>>(out, 1.f / (float)(kNT * kOPD));
}
